// MSIB_Extractor_GIN_57724360458774
// MI455X (gfx1250) — matrix operand order repaired and checked by reading, not compiled, not run on hardware
//
#include <hip/hip_runtime.h>
#include <hip/hip_bf16.h>
#include <math.h>

#define N_NODES 100000
#define N_EDGES 1600000
#define DIM 128

typedef __attribute__((ext_vector_type(16))) _Float16 v16h;
typedef __attribute__((ext_vector_type(8)))  float    v8f;

union Frag { unsigned u[8]; v16h h; };

// ---------------------------------------------------------------------------
// Weight prep: f32 [k][n] -> f16 transposed [n][k] (so B-fragment K-pairs are
// contiguous dwords in LDS).
// ---------------------------------------------------------------------------
__global__ __launch_bounds__(256) void prep_weights(const float* __restrict__ W,
                                                    _Float16* __restrict__ WT) {
  int tid = blockIdx.x * 256 + threadIdx.x;          // 16384 elements
  if (tid < DIM * DIM) {
    int n = tid >> 7;        // output channel
    int k = tid & 127;       // input channel
    WT[n * DIM + k] = (_Float16)W[k * DIM + n];
  }
}

// ---------------------------------------------------------------------------
// agg := x  (covers the "+ x_i" self term of GIN with eps=0)
// ---------------------------------------------------------------------------
__global__ __launch_bounds__(256) void copy_f32(const float* __restrict__ src,
                                                float* __restrict__ dst, long n4) {
  long i = (long)blockIdx.x * 256 + threadIdx.x;
  if (i < n4) ((float4*)dst)[i] = ((const float4*)src)[i];
}

// ---------------------------------------------------------------------------
// agg[dst] += x[src] : one wave per edge -> coalesced 512B row read,
// 4 global_atomic_add_f32 per lane.
// ---------------------------------------------------------------------------
__global__ __launch_bounds__(256) void scatter_add(const float* __restrict__ x,
                                                   float* __restrict__ agg,
                                                   const int* __restrict__ ei,
                                                   int n_edges) {
  long gid = (long)blockIdx.x * 256 + threadIdx.x;
  int edge = (int)(gid >> 5);
  int lane = (int)(gid & 31);
  if (edge >= n_edges) return;
  int s = ei[edge];
  int d = ei[n_edges + edge];
  float4 v = ((const float4*)(x + (size_t)s * DIM))[lane];
  float* p = agg + (size_t)d * DIM + lane * 4;
  unsafeAtomicAdd(p + 0, v.x);
  unsafeAtomicAdd(p + 1, v.y);
  unsafeAtomicAdd(p + 2, v.z);
  unsafeAtomicAdd(p + 3, v.w);
}

// ---------------------------------------------------------------------------
// Per-wave 16x128 strip GEMM against a 128x128 LDS weight (stored [n][k]).
// Fragment packing per CDNA5 ISA 7.12.2 (wave32, 16-bit A 16x32, B 32x16).
// ---------------------------------------------------------------------------
__device__ __forceinline__ void wave_gemm_16x128(const _Float16* A,   // LDS 16x128
                                                 const _Float16* WT,  // LDS 128x128 [n][k]
                                                 const float* __restrict__ bias,
                                                 v8f acc[8], int lane) {
  const int mlo    = lane & 15;
  const int khalfA = (lane & 16) ? 8  : 0;   // A: lanes 16-31 hold K+8 pairs
  const int khalfB = (lane & 16) ? 8  : 0;   // B: lanes 16-31 hold K+8 pairs
#pragma unroll
  for (int nt = 0; nt < 8; ++nt) acc[nt] = (v8f){0.f,0.f,0.f,0.f,0.f,0.f,0.f,0.f};

  for (int k0 = 0; k0 < DIM; k0 += 32) {
    Frag a;
#pragma unroll
    for (int i = 0; i < 8; ++i) {
      int k = k0 + khalfA + ((i < 4) ? 2 * i : 16 + 2 * (i - 4));
      a.u[i] = *(const unsigned*)(A + mlo * DIM + k);
    }
#pragma unroll
    for (int nt = 0; nt < 8; ++nt) {
      Frag b;
      int n = nt * 16 + mlo;
#pragma unroll
      for (int v = 0; v < 8; ++v) {
        int k = k0 + khalfB + ((v < 4) ? 2 * v : 16 + 2 * (v - 4));
        b.u[v] = *(const unsigned*)(WT + n * DIM + k);
      }
      acc[nt] = __builtin_amdgcn_wmma_f32_16x16x32_f16(
          false, a.h, false, b.h, (short)0, acc[nt], false, false);
    }
  }
#pragma unroll
  for (int nt = 0; nt < 8; ++nt) {
    float bv = bias[nt * 16 + mlo];
#pragma unroll
    for (int r = 0; r < 8; ++r) acc[nt][r] += bv;
  }
}

// ---------------------------------------------------------------------------
// Fused GIN MLP: h := [relu]( relu(h @ W1 + b1) @ W2 + b2 ), in place.
// 128 rows per block, 8 waves, intermediate lives in LDS only.
// ---------------------------------------------------------------------------
__global__ __launch_bounds__(256) void gin_mlp(float* __restrict__ buf,
                                               const _Float16* __restrict__ WT1,
                                               const float* __restrict__ b1,
                                               const _Float16* __restrict__ WT2,
                                               const float* __restrict__ b2,
                                               int n_nodes, int relu_out) {
  __shared__ _Float16 sA [128 * DIM];   // 32 KB activation strip (wave-private 16-row slices)
  __shared__ _Float16 sW1[DIM * DIM];   // 32 KB
  __shared__ _Float16 sW2[DIM * DIM];   // 32 KB

  const int tid  = threadIdx.x;
  const int wave = tid >> 5;
  const int lane = tid & 31;

  // cooperative weight load (f16, already transposed)
  const uint2* g1 = (const uint2*)WT1;
  const uint2* g2 = (const uint2*)WT2;
  uint2* s1 = (uint2*)sW1;
  uint2* s2 = (uint2*)sW2;
  for (int i = tid; i < DIM * DIM / 4; i += 256) { s1[i] = g1[i]; s2[i] = g2[i]; }

  // load this wave's 16 rows, f32 -> f16 (clamp row so EXEC stays full for WMMA)
  const int rowBase = blockIdx.x * 128 + wave * 16;
  _Float16* A = sA + wave * 16 * DIM;
  for (int r = 0; r < 16; ++r) {
    int gr = rowBase + r; if (gr >= n_nodes) gr = n_nodes - 1;
    float4 v = ((const float4*)(buf + (size_t)gr * DIM))[lane];
    _Float16* p = A + r * DIM + lane * 4;
    p[0] = (_Float16)v.x; p[1] = (_Float16)v.y; p[2] = (_Float16)v.z; p[3] = (_Float16)v.w;
  }
  __syncthreads();

  v8f acc[8];
  const int mlo = lane & 15;
  const int mhi = (lane & 16) ? 8 : 0;

  // GEMM1 + bias + ReLU -> back into LDS strip (wave-private, in place)
  wave_gemm_16x128(A, sW1, b1, acc, lane);
#pragma unroll
  for (int nt = 0; nt < 8; ++nt) {
    int n = nt * 16 + mlo;
#pragma unroll
    for (int r = 0; r < 8; ++r) {
      float v = fmaxf(acc[nt][r], 0.0f);
      A[(r + mhi) * DIM + n] = (_Float16)v;
    }
  }

  // GEMM2 + bias (+ optional ReLU) -> global, in place
  wave_gemm_16x128(A, sW2, b2, acc, lane);
#pragma unroll
  for (int nt = 0; nt < 8; ++nt) {
    int n = nt * 16 + mlo;
#pragma unroll
    for (int r = 0; r < 8; ++r) {
      float v = acc[nt][r];
      if (relu_out) v = fmaxf(v, 0.0f);
      int gm = rowBase + r + mhi;
      if (gm < n_nodes) buf[(size_t)gm * DIM + n] = v;
    }
  }
}

// ---------------------------------------------------------------------------
// Readout: sigmoid(concat(h0,h1,h2) @ Wm + bm). One wave per node.
// ---------------------------------------------------------------------------
__global__ __launch_bounds__(256) void readout(const float* __restrict__ h0,
                                               const float* __restrict__ h1,
                                               const float* __restrict__ h2,
                                               const float* __restrict__ Wm,
                                               const float* __restrict__ bm,
                                               float* __restrict__ out, int n_nodes) {
  long gid = (long)blockIdx.x * 256 + threadIdx.x;
  int node = (int)(gid >> 5);
  int lane = (int)(gid & 31);
  if (node >= n_nodes) return;
  const float* bufs[3] = {h0, h1, h2};
  float sum = 0.f;
#pragma unroll
  for (int i = 0; i < 12; ++i) {
    int k = lane + 32 * i;                       // 0..383
    sum += bufs[k >> 7][(size_t)node * DIM + (k & 127)] * Wm[k];
  }
#pragma unroll
  for (int off = 16; off > 0; off >>= 1) sum += __shfl_down(sum, off, 32);
  if (lane == 0) out[node] = 1.0f / (1.0f + __expf(-(sum + bm[0])));
}

// ---------------------------------------------------------------------------
extern "C" void kernel_launch(void* const* d_in, const int* in_sizes, int n_in,
                              void* d_out, int out_size, void* d_ws, size_t ws_size,
                              hipStream_t stream) {
  const float* x  = (const float*)d_in[0];
  const int*   ei = (const int*)d_in[1];
  const float* W1[3] = {(const float*)d_in[3],  (const float*)d_in[7],  (const float*)d_in[11]};
  const float* b1[3] = {(const float*)d_in[4],  (const float*)d_in[8],  (const float*)d_in[12]};
  const float* W2[3] = {(const float*)d_in[5],  (const float*)d_in[9],  (const float*)d_in[13]};
  const float* b2[3] = {(const float*)d_in[6],  (const float*)d_in[10], (const float*)d_in[14]};
  const float* Wm = (const float*)d_in[15];
  const float* bm = (const float*)d_in[16];

  char* ws = (char*)d_ws;
  const size_t bufBytes = (size_t)N_NODES * DIM * sizeof(float);   // 51.2 MB
  float* h[3] = {(float*)ws, (float*)(ws + bufBytes), (float*)(ws + 2 * bufBytes)};
  _Float16* wts = (_Float16*)(ws + 3 * bufBytes);                  // 6 x 32 KB f16 weights

  // convert + transpose all weights once
  for (int i = 0; i < 3; ++i) {
    prep_weights<<<64, 256, 0, stream>>>(W1[i], wts + (2 * i + 0) * DIM * DIM);
    prep_weights<<<64, 256, 0, stream>>>(W2[i], wts + (2 * i + 1) * DIM * DIM);
  }

  const long n4 = (long)N_NODES * DIM / 4;                 // 3.2M float4
  const int copyBlocks = (int)((n4 + 255) / 256);
  const int scatBlocks = (int)(((long)N_EDGES * 32 + 255) / 256);  // 200000
  const int mlpBlocks  = (N_NODES + 127) / 128;                    // 782

  const float* cur = x;
  for (int i = 0; i < 3; ++i) {
    copy_f32   <<<copyBlocks, 256, 0, stream>>>(cur, h[i], n4);
    scatter_add<<<scatBlocks, 256, 0, stream>>>(cur, h[i], ei, N_EDGES);
    gin_mlp    <<<mlpBlocks, 256, 0, stream>>>(h[i],
                                               wts + (2 * i + 0) * DIM * DIM, b1[i],
                                               wts + (2 * i + 1) * DIM * DIM, b2[i],
                                               N_NODES, (i < 2) ? 1 : 0);
    cur = h[i];
  }

  const int roBlocks = (int)(((long)N_NODES * 32 + 255) / 256);
  readout<<<roBlocks, 256, 0, stream>>>(h[0], h[1], h[2], Wm, bm, (float*)d_out, N_NODES);
}
